// MixGNN_88613765251902
// MI455X (gfx1250) — matrix operand order repaired and checked by reading, not compiled, not run on hardware
//
#include <hip/hip_runtime.h>

// ---------------------------------------------------------------------------
// MixGNN forward for MI455X (gfx1250), wave32 + WMMA bf16.
//
// Input flattening assumption (recursive dict insertion order):
//  0:s 1:v 2:p 3:edge_attr
//  4..11  pre   conv {We1,be1,We2,be2,Wu1,bu1,Wu2,bu2}
//  12..19 conv0, 20..27 conv1, 28..35 conv2, 36..43 post
//  44..52 norms[0..2]{g,b,gv}, 53..55 pnorm0, 56..58 pnorm1
//  59:ep1W 60:ep1b 61:ep2W 62:ep2b 63:eq1W 64:eq1b 65:eq2W 66:eq2b
//  67:edge_index (2xE) 68:batch (unused)
// Output: concat(s[2048x256], v[2048x192], e[129024x16], p[2048x3]) f32.
// ---------------------------------------------------------------------------

#define SDIM   256
#define VDIM   64
#define EDIM   16
#define NNODE  2048
#define NPG    64        // nodes per graph
#define NEDGE  129024
#define EPG    4032      // edges per graph = 64*63
#define KE1    544       // padded in_e (530 -> 544)
#define OEPAD  400       // padded out_e (385 -> 400)

typedef __attribute__((ext_vector_type(16))) __bf16 bf16x16;
typedef __attribute__((ext_vector_type(8)))  float  f32x8;

struct alignas(16) I4 { int x, y, z, w; };
union FragU { bf16x16 v; I4 q[2]; };

__device__ __forceinline__ float silu_f(float x) { return x / (1.0f + __expf(-x)); }
__device__ __forceinline__ f32x8 zero8() {
  f32x8 z = {0.f, 0.f, 0.f, 0.f, 0.f, 0.f, 0.f, 0.f};
  return z;
}

// A fragment (16x32 bf16, MxK) from row-major LDS tile, stride in elements.
// Lane L: row = L%16; K = base..base+7 and base+16..base+23, base = (L<16?0:8).
__device__ __forceinline__ bf16x16 load_a_frag(const __bf16* base, int stride, int lane) {
  const int row = lane & 15;
  const int kb  = (lane & 16) ? 8 : 0;
  const __bf16* p = base + row * stride + kb;
  FragU f;
  f.q[0] = *(const I4*)(p);
  f.q[1] = *(const I4*)(p + 16);
  return f.v;
}

// B fragment (32x16 bf16, KxN) from weight pre-transposed as WT[n][Kpad].
// Lane L: col = L%16; K = base..base+7 and base+16..base+23, base = (L<16?0:8).
__device__ __forceinline__ bf16x16 load_b_frag(const __bf16* WT, int n0, int Kpad,
                                               int kc, int lane) {
  const __bf16* p = WT + (size_t)(n0 + (lane & 15)) * Kpad + kc + ((lane & 16) ? 8 : 0);
  FragU f;
  f.q[0] = *(const I4*)(p);
  f.q[1] = *(const I4*)(p + 16);
  return f.v;
}

__device__ __forceinline__ f32x8 wmma_bf16(bf16x16 a, bf16x16 b, f32x8 c) {
  return __builtin_amdgcn_wmma_f32_16x16x32_bf16(false, a, false, b, (short)0, c,
                                                 false, false);
}

// ---------------------------------------------------------------------------
// Weight prep: W (fi x fo, row-major f32) -> WT (fopad x Kpad, bf16), zero pad.
// ---------------------------------------------------------------------------
__global__ void prep_weight(const float* __restrict__ W, __bf16* __restrict__ WT,
                            int fi, int fo, int Kpad, int fopad) {
  int idx = blockIdx.x * blockDim.x + threadIdx.x;
  int total = fopad * Kpad;
  if (idx >= total) return;
  int n = idx / Kpad, k = idx % Kpad;
  float v = (n < fo && k < fi) ? W[(size_t)k * fo + n] : 0.f;
  WT[idx] = (__bf16)v;
}

__global__ void convert_bf16(const float* __restrict__ src, __bf16* __restrict__ dst, int n) {
  int i = blockIdx.x * blockDim.x + threadIdx.x;
  if (i < n) dst[i] = (__bf16)src[i];
}

// ---------------------------------------------------------------------------
// Fused edge-MLP + segment-mean aggregation. One block per target node.
// 128 threads = 4 waves; wave w owns edge rows [16w, 16w+16).
// ---------------------------------------------------------------------------
__global__ __launch_bounds__(128) void conv_edge_agg(
    const float* __restrict__ pos, const float* __restrict__ vcur,
    const __bf16* __restrict__ sbf, const float* __restrict__ eattr,
    const __bf16* __restrict__ We1T, const float* __restrict__ be1,
    const __bf16* __restrict__ We2T, const float* __restrict__ be2,
    __bf16* __restrict__ saggbf, float* __restrict__ vagg, float* __restrict__ pagg,
    int has_v, float cutoff) {
  __shared__ float shP[NPG][3], shPN[NPG][3], shRN[NPG][3];
  __shared__ float shW[NPG], shWC[NPG], shD[NPG], shANG[NPG];
  __shared__ __bf16 shA[NPG][40];    // K-chunk staging (32 + pad 8)
  __shared__ __bf16 shH[NPG][264];   // hidden activations (256 + pad 8)
  __shared__ float shGVR[NPG][2 * VDIM];  // [m][0..63]=g_v*w, [m][64..127]=g_r*w
  __shared__ float shSacc[SDIM];
  __shared__ float shGP[NPG];
  __shared__ float shInvDen;

  const int tid  = threadIdx.x;
  const int t    = blockIdx.x;  // target node
  const int g    = t >> 6;
  const int jj   = t & 63;
  const int lane = tid & 31;
  const int wave = tid >> 5;
  const int m0   = wave * 16;

  // ---- phase 0: geometry ----
  if (tid < NPG) {
    int node = g * NPG + tid;
    float px = pos[node * 3 + 0], py = pos[node * 3 + 1], pz = pos[node * 3 + 2];
    shP[tid][0] = px; shP[tid][1] = py; shP[tid][2] = pz;
    float inv = rsqrtf(px * px + py * py + pz * pz);
    shPN[tid][0] = px * inv; shPN[tid][1] = py * inv; shPN[tid][2] = pz * inv;
  }
  for (int i = tid; i < SDIM; i += 128) shSacc[i] = 0.f;
  __syncthreads();
  if (tid < NPG) {
    int m = tid;  // source node index within graph
    float rx = shP[jj][0] - shP[m][0];
    float ry = shP[jj][1] - shP[m][1];
    float rz = shP[jj][2] - shP[m][2];
    float d  = sqrtf(fmaxf(rx * rx + ry * ry + rz * rz, 1e-6f));
    float ad = shPN[jj][0] * shPN[m][0] + shPN[jj][1] * shPN[m][1] +
               shPN[jj][2] * shPN[m][2];
    float sc = 1.0f / (1.0f + d);
    float w, wc;
    if (cutoff > 0.f) {
      float inside = (d <= cutoff) ? 1.f : 0.f;
      float x = fminf(d / cutoff, 1.f);
      w  = 0.5f * (cosf(3.14159265358979323846f * x) + 1.f) * inside;
      wc = inside;
    } else { w = 1.f; wc = 1.f; }
    if (m == jj) { w = 0.f; wc = 0.f; rx = ry = rz = 0.f; d = 0.f; ad = 0.f; sc = 0.f; }
    shRN[m][0] = rx * sc; shRN[m][1] = ry * sc; shRN[m][2] = rz * sc;
    shW[m] = w; shWC[m] = wc; shD[m] = d; shANG[m] = ad;
  }
  __syncthreads();
  if (tid == 0) {
    float den = 0.f;
    for (int m = 0; m < NPG; ++m) den += shWC[m];
    shInvDen = 1.f / fmaxf(den, 1.f);
  }

  // ---- phase 1: h = silu(feat @ We1 + be1), feat = [s_src | s_tgt | d,a,e] ----
  f32x8 acc[16];
  #pragma unroll
  for (int i = 0; i < 16; ++i) acc[i] = zero8();

  for (int kc = 0; kc < 17; ++kc) {
    if (kc < 16) {
      int m = tid >> 1, half = tid & 1;
      int node = (kc < 8) ? (g * NPG + m) : t;
      int col  = ((kc & 7) << 5) + half * 16;
      const I4* sp = (const I4*)(sbf + (size_t)node * SDIM + col);
      I4* dp = (I4*)(&shA[m][half * 16]);
      dp[0] = sp[0]; dp[1] = sp[1];
    } else if (tid < NPG) {
      int m = tid;
      shA[m][0] = (__bf16)shD[m];
      shA[m][1] = (__bf16)shANG[m];
      if (m != jj) {
        int eid = g * EPG + m * 63 + jj - ((jj > m) ? 1 : 0);
        #pragma unroll
        for (int k = 0; k < EDIM; ++k)
          shA[m][2 + k] = (__bf16)eattr[(size_t)eid * EDIM + k];
      } else {
        #pragma unroll
        for (int k = 0; k < EDIM; ++k) shA[m][2 + k] = (__bf16)0.f;
      }
      for (int k = 2 + EDIM; k < 32; ++k) shA[m][k] = (__bf16)0.f;
    }
    __syncthreads();
    bf16x16 af = load_a_frag(&shA[m0][0], 40, lane);
    #pragma unroll
    for (int nt = 0; nt < 16; ++nt) {
      bf16x16 bfr = load_b_frag(We1T, nt * 16, KE1, kc * 32, lane);
      acc[nt] = wmma_bf16(af, bfr, acc[nt]);
    }
    __syncthreads();
  }
  {
    int hi8 = (lane & 16) ? 8 : 0, nl = lane & 15;
    #pragma unroll
    for (int nt = 0; nt < 16; ++nt) {
      int n = nt * 16 + nl;
      float bias = be1[n];
      #pragma unroll
      for (int r = 0; r < 8; ++r)
        shH[m0 + r + hi8][n] = (__bf16)silu_f(acc[nt][r] + bias);
    }
  }
  __syncthreads();

  // ---- phase 2: out = h @ We2 + be2, fused message weighting + reduction ----
  for (int nt = 0; nt < 25; ++nt) {
    f32x8 c = zero8();
    #pragma unroll
    for (int kc = 0; kc < 8; ++kc) {
      bf16x16 af  = load_a_frag(&shH[m0][kc * 32], 264, lane);
      bf16x16 bfr = load_b_frag(We2T, nt * 16, SDIM, kc * 32, lane);
      c = wmma_bf16(af, bfr, c);
    }
    int hi8 = (lane & 16) ? 8 : 0;
    int n = nt * 16 + (lane & 15);
    if (n < SDIM) {                      // m_s: weighted column-sum over edges
      float bias = be2[n], partial = 0.f;
      #pragma unroll
      for (int r = 0; r < 8; ++r) partial += (c[r] + bias) * shW[m0 + r + hi8];
      atomicAdd(&shSacc[n], partial);
    } else if (n < SDIM + 2 * VDIM) {    // g_v / g_r, weight folded in
      float bias = be2[n];
      #pragma unroll
      for (int r = 0; r < 8; ++r) {
        int m = m0 + r + hi8;
        shGVR[m][n - SDIM] = (c[r] + bias) * shW[m];
      }
    } else if (n == SDIM + 2 * VDIM) {   // g_p
      float bias = be2[n];
      #pragma unroll
      for (int r = 0; r < 8; ++r) {
        int m = m0 + r + hi8;
        shGP[m] = tanhf(c[r] + bias) * shW[m];
      }
    }
  }
  __syncthreads();

  // ---- phase 3: normalize by count and write aggregates ----
  float invden = shInvDen;
  for (int n = tid; n < SDIM; n += 128)
    saggbf[(size_t)t * SDIM + n] = (__bf16)(shSacc[n] * invden);
  for (int idx = tid; idx < 3 * VDIM; idx += 128) {
    int dim = idx / VDIM, ch = idx % VDIM;
    float sum = 0.f;
    for (int m = 0; m < NPG; ++m) {
      float val = shGVR[m][VDIM + ch] * shRN[m][dim];
      if (has_v)
        val += shGVR[m][ch] *
               vcur[(size_t)(g * NPG + m) * (3 * VDIM) + dim * VDIM + ch];
      sum += val;
    }
    vagg[(size_t)t * (3 * VDIM) + dim * VDIM + ch] = sum * invden;
  }
  if (tid < 3) {
    float sum = 0.f;
    for (int m = 0; m < NPG; ++m) sum += shGP[m] * shRN[m][tid];
    pagg[t * 3 + tid] = sum * invden;
  }
}

// ---------------------------------------------------------------------------
// Node update: s += silu([s|s_agg]@Wu1+bu1)@Wu2+bu2 ; v += v_agg ; p += p_agg.
// One block per 64 nodes, 128 threads.
// ---------------------------------------------------------------------------
__global__ __launch_bounds__(128) void node_update(
    const float* __restrict__ s_in, const __bf16* __restrict__ sbf,
    const __bf16* __restrict__ saggbf,
    const __bf16* __restrict__ Wu1T, const float* __restrict__ bu1,
    const __bf16* __restrict__ Wu2T, const float* __restrict__ bu2,
    float* __restrict__ s_out,
    const float* __restrict__ v_in, const float* __restrict__ vagg,
    float* __restrict__ v_out,
    const float* __restrict__ p_in, const float* __restrict__ pagg,
    float* __restrict__ p_out) {
  __shared__ __bf16 shA[64][40];
  __shared__ __bf16 shH[64][264];
  const int tid = threadIdx.x, lane = tid & 31, wave = tid >> 5;
  const int nb = blockIdx.x * 64;
  const int m0 = wave * 16;

  for (int i = tid; i < 64 * 3 * VDIM; i += 128) {
    size_t idx = (size_t)nb * 3 * VDIM + i;
    v_out[idx] = v_in[idx] + vagg[idx];
  }
  for (int i = tid; i < 64 * 3; i += 128) {
    size_t idx = (size_t)nb * 3 + i;
    p_out[idx] = p_in[idx] + pagg[idx];
  }

  f32x8 acc[16];
  #pragma unroll
  for (int i = 0; i < 16; ++i) acc[i] = zero8();
  for (int kc = 0; kc < 16; ++kc) {
    int m = tid >> 1, half = tid & 1;
    const __bf16* srcb = (kc < 8) ? sbf : saggbf;
    int col = ((kc & 7) << 5) + half * 16;
    const I4* sp = (const I4*)(srcb + (size_t)(nb + m) * SDIM + col);
    I4* dp = (I4*)(&shA[m][half * 16]);
    dp[0] = sp[0]; dp[1] = sp[1];
    __syncthreads();
    bf16x16 af = load_a_frag(&shA[m0][0], 40, lane);
    #pragma unroll
    for (int nt = 0; nt < 16; ++nt) {
      bf16x16 bfr = load_b_frag(Wu1T, nt * 16, 2 * SDIM, kc * 32, lane);
      acc[nt] = wmma_bf16(af, bfr, acc[nt]);
    }
    __syncthreads();
  }
  {
    int hi8 = (lane & 16) ? 8 : 0, nl = lane & 15;
    #pragma unroll
    for (int nt = 0; nt < 16; ++nt) {
      int n = nt * 16 + nl;
      float bias = bu1[n];
      #pragma unroll
      for (int r = 0; r < 8; ++r)
        shH[m0 + r + hi8][n] = (__bf16)silu_f(acc[nt][r] + bias);
    }
  }
  __syncthreads();

  f32x8 acc2[16];
  #pragma unroll
  for (int i = 0; i < 16; ++i) acc2[i] = zero8();
  #pragma unroll
  for (int kc = 0; kc < 8; ++kc) {
    bf16x16 af = load_a_frag(&shH[m0][kc * 32], 264, lane);
    #pragma unroll
    for (int nt = 0; nt < 16; ++nt) {
      bf16x16 bfr = load_b_frag(Wu2T, nt * 16, SDIM, kc * 32, lane);
      acc2[nt] = wmma_bf16(af, bfr, acc2[nt]);
    }
  }
  {
    int hi8 = (lane & 16) ? 8 : 0, nl = lane & 15;
    #pragma unroll
    for (int nt = 0; nt < 16; ++nt) {
      int n = nt * 16 + nl;
      float bias = bu2[n];
      #pragma unroll
      for (int r = 0; r < 8; ++r) {
        size_t node = nb + m0 + r + hi8;
        s_out[node * SDIM + n] = s_in[node * SDIM + n] + acc2[nt][r] + bias;
      }
    }
  }
}

// ---------------------------------------------------------------------------
// LayerNorm on s + vector-norm on v; also emits bf16 copy of s.
// One block (256 threads) per node.
// ---------------------------------------------------------------------------
__global__ __launch_bounds__(256) void lnorm_kernel(
    const float* __restrict__ s_in, const float* __restrict__ v_in,
    const float* __restrict__ gg, const float* __restrict__ bb,
    const float* __restrict__ gv,
    float* __restrict__ s_out, float* __restrict__ v_out,
    __bf16* __restrict__ sbf_out) {
  __shared__ float red[256];
  const int tid = threadIdx.x, t = blockIdx.x;
  float x = s_in[(size_t)t * SDIM + tid];

  red[tid] = x; __syncthreads();
  for (int s = 128; s > 0; s >>= 1) { if (tid < s) red[tid] += red[tid + s]; __syncthreads(); }
  float mu = red[0] / SDIM; __syncthreads();

  float dx = x - mu;
  red[tid] = dx * dx; __syncthreads();
  for (int s = 128; s > 0; s >>= 1) { if (tid < s) red[tid] += red[tid + s]; __syncthreads(); }
  float var = red[0] / SDIM; __syncthreads();

  float sn = dx * rsqrtf(var + 1e-5f) * gg[tid] + bb[tid];
  s_out[(size_t)t * SDIM + tid] = sn;
  sbf_out[(size_t)t * SDIM + tid] = (__bf16)sn;

  float vv = 0.f, y = 0.f;
  if (tid < 3 * VDIM) { y = v_in[(size_t)t * 3 * VDIM + tid]; vv = y * y; }
  red[tid] = vv; __syncthreads();
  for (int s = 128; s > 0; s >>= 1) { if (tid < s) red[tid] += red[tid + s]; __syncthreads(); }
  float vn2 = red[0] / VDIM;
  float scale = rsqrtf(vn2 + 1e-5f);
  if (tid < 3 * VDIM)
    v_out[(size_t)t * 3 * VDIM + tid] = y * scale * gv[tid % VDIM];
}

// ---------------------------------------------------------------------------
// en = silu(s@ep1W+ep1b)@ep2W+ep2b  (N x 16). One block per 64 nodes.
// ---------------------------------------------------------------------------
__global__ __launch_bounds__(128) void en_mlp(
    const __bf16* __restrict__ sbf, const __bf16* __restrict__ ep1T,
    const float* __restrict__ ep1b, const __bf16* __restrict__ ep2T,
    const float* __restrict__ ep2b, float* __restrict__ en) {
  __shared__ __bf16 shA[64][40];
  __shared__ __bf16 shH[64][264];
  const int tid = threadIdx.x, lane = tid & 31, wave = tid >> 5;
  const int nb = blockIdx.x * 64;
  const int m0 = wave * 16;

  f32x8 acc[16];
  #pragma unroll
  for (int i = 0; i < 16; ++i) acc[i] = zero8();
  for (int kc = 0; kc < 8; ++kc) {
    int m = tid >> 1, half = tid & 1;
    int col = (kc << 5) + half * 16;
    const I4* sp = (const I4*)(sbf + (size_t)(nb + m) * SDIM + col);
    I4* dp = (I4*)(&shA[m][half * 16]);
    dp[0] = sp[0]; dp[1] = sp[1];
    __syncthreads();
    bf16x16 af = load_a_frag(&shA[m0][0], 40, lane);
    #pragma unroll
    for (int nt = 0; nt < 16; ++nt) {
      bf16x16 bfr = load_b_frag(ep1T, nt * 16, SDIM, kc * 32, lane);
      acc[nt] = wmma_bf16(af, bfr, acc[nt]);
    }
    __syncthreads();
  }
  {
    int hi8 = (lane & 16) ? 8 : 0, nl = lane & 15;
    #pragma unroll
    for (int nt = 0; nt < 16; ++nt) {
      int n = nt * 16 + nl;
      float bias = ep1b[n];
      #pragma unroll
      for (int r = 0; r < 8; ++r)
        shH[m0 + r + hi8][n] = (__bf16)silu_f(acc[nt][r] + bias);
    }
  }
  __syncthreads();

  f32x8 c = zero8();
  #pragma unroll
  for (int kc = 0; kc < 8; ++kc) {
    bf16x16 af  = load_a_frag(&shH[m0][kc * 32], 264, lane);
    bf16x16 bfr = load_b_frag(ep2T, 0, SDIM, kc * 32, lane);
    c = wmma_bf16(af, bfr, c);
  }
  {
    int hi8 = (lane & 16) ? 8 : 0;
    int n = lane & 15;
    float bias = ep2b[n];
    #pragma unroll
    for (int r = 0; r < 8; ++r)
      en[(size_t)(nb + m0 + r + hi8) * EDIM + n] = c[r] + bias;
  }
}

// ---------------------------------------------------------------------------
// Final edge head: e = silu((eattr + en[src] + en[tgt])@eq1W+b)@eq2W+b.
// ---------------------------------------------------------------------------
__global__ __launch_bounds__(256) void edge_out(
    const float* __restrict__ eattr, const float* __restrict__ en,
    const int* __restrict__ ei,
    const float* __restrict__ q1W, const float* __restrict__ q1b,
    const float* __restrict__ q2W, const float* __restrict__ q2b,
    float* __restrict__ eout) {
  __shared__ float W1[16][16], W2[16][16], B1[16], B2[16];
  const int tid = threadIdx.x;
  W1[tid >> 4][tid & 15] = q1W[tid];
  W2[tid >> 4][tid & 15] = q2W[tid];
  if (tid < 16) { B1[tid] = q1b[tid]; B2[tid] = q2b[tid]; }
  __syncthreads();
  int e = blockIdx.x * 256 + tid;
  if (e >= NEDGE) return;
  int src = ei[e], tgt = ei[NEDGE + e];
  float tf[16], h[16];
  #pragma unroll
  for (int k = 0; k < 16; ++k)
    tf[k] = eattr[(size_t)e * 16 + k] + en[(size_t)src * 16 + k] + en[(size_t)tgt * 16 + k];
  #pragma unroll
  for (int j = 0; j < 16; ++j) {
    float s = B1[j];
    #pragma unroll
    for (int k = 0; k < 16; ++k) s += tf[k] * W1[k][j];
    h[j] = silu_f(s);
  }
  #pragma unroll
  for (int j = 0; j < 16; ++j) {
    float s = B2[j];
    #pragma unroll
    for (int k = 0; k < 16; ++k) s += h[k] * W2[k][j];
    eout[(size_t)e * 16 + j] = s;
  }
}

// ---------------------------------------------------------------------------
// Host orchestration (graph-capture safe: launches only).
// ---------------------------------------------------------------------------
extern "C" void kernel_launch(void* const* d_in, const int* in_sizes, int n_in,
                              void* d_out, int out_size, void* d_ws, size_t ws_size,
                              hipStream_t stream) {
  (void)in_sizes; (void)n_in; (void)out_size; (void)ws_size;
  auto F = [&](int i) { return (const float*)d_in[i]; };
  const float* in_s  = F(0);
  const float* in_v  = F(1);
  const float* in_p  = F(2);
  const float* eattr = F(3);
  const int*   ei    = (const int*)d_in[67];

  float* out   = (float*)d_out;
  float* out_s = out;
  float* out_v = out + (size_t)NNODE * SDIM;
  float* out_e = out_v + (size_t)NNODE * 3 * VDIM;
  float* out_p = out_e + (size_t)NEDGE * EDIM;

  char* ws = (char*)d_ws;
  size_t off = 0;
  auto alloc = [&](size_t bytes) {
    char* p = ws + off;
    off = (off + bytes + 255) & ~(size_t)255;
    return p;
  };
  __bf16* sbf    = (__bf16*)alloc((size_t)NNODE * SDIM * 2);
  __bf16* saggbf = (__bf16*)alloc((size_t)NNODE * SDIM * 2);
  float*  s_cur  = (float*)alloc((size_t)NNODE * SDIM * 4);
  float*  v_cur  = (float*)alloc((size_t)NNODE * 3 * VDIM * 4);
  float*  p_cur  = (float*)alloc((size_t)NNODE * 3 * 4);
  float*  vagg   = (float*)alloc((size_t)NNODE * 3 * VDIM * 4);
  float*  pagg   = (float*)alloc((size_t)NNODE * 3 * 4);
  float*  en     = (float*)alloc((size_t)NNODE * EDIM * 4);
  __bf16* we1t   = (__bf16*)alloc((size_t)5 * SDIM * KE1 * 2);
  __bf16* we2t   = (__bf16*)alloc((size_t)5 * OEPAD * SDIM * 2);
  __bf16* wu1t   = (__bf16*)alloc((size_t)5 * SDIM * 2 * SDIM * 2);
  __bf16* wu2t   = (__bf16*)alloc((size_t)5 * SDIM * SDIM * 2);
  __bf16* ep1t   = (__bf16*)alloc((size_t)SDIM * SDIM * 2);
  __bf16* ep2t   = (__bf16*)alloc((size_t)EDIM * SDIM * 2);

  static const int CB[5] = {4, 12, 20, 28, 36};   // pre, conv0..2, post
  static const int NB[5] = {44, 47, 50, 53, 56};  // norms0..2, pnorm0, pnorm1

  auto prep = [&](const float* W, __bf16* WT, int fi, int fo, int Kpad, int fopad) {
    int total = fopad * Kpad;
    prep_weight<<<(total + 255) / 256, 256, 0, stream>>>(W, WT, fi, fo, Kpad, fopad);
  };
  for (int L = 0; L < 5; ++L) {
    prep(F(CB[L] + 0), we1t + (size_t)L * SDIM * KE1,      530, SDIM, KE1, SDIM);
    prep(F(CB[L] + 2), we2t + (size_t)L * OEPAD * SDIM,    SDIM, 385, SDIM, OEPAD);
    prep(F(CB[L] + 4), wu1t + (size_t)L * SDIM * 2 * SDIM, 2 * SDIM, SDIM, 2 * SDIM, SDIM);
    prep(F(CB[L] + 6), wu2t + (size_t)L * SDIM * SDIM,     SDIM, SDIM, SDIM, SDIM);
  }
  prep(F(59), ep1t, SDIM, SDIM, SDIM, SDIM);
  prep(F(61), ep2t, SDIM, EDIM, SDIM, EDIM);

  convert_bf16<<<(NNODE * SDIM) / 256, 256, 0, stream>>>(in_s, sbf, NNODE * SDIM);

  auto run_conv = [&](int L, const float* pos, const float* vc, int has_v, float cut) {
    conv_edge_agg<<<NNODE, 128, 0, stream>>>(
        pos, vc, sbf, eattr, we1t + (size_t)L * SDIM * KE1, F(CB[L] + 1),
        we2t + (size_t)L * OEPAD * SDIM, F(CB[L] + 3), saggbf, vagg, pagg, has_v, cut);
  };
  auto run_update = [&](int L, const float* si, float* so, const float* vi, float* vo,
                        const float* pi, float* po) {
    node_update<<<NNODE / 64, 128, 0, stream>>>(
        si, sbf, saggbf, wu1t + (size_t)L * SDIM * 2 * SDIM, F(CB[L] + 5),
        wu2t + (size_t)L * SDIM * SDIM, F(CB[L] + 7), so, vi, vagg, vo, pi, pagg, po);
  };
  auto run_norm = [&](int Nn, float* so, float* vo) {
    lnorm_kernel<<<NNODE, 256, 0, stream>>>(s_cur, v_cur, F(NB[Nn] + 0), F(NB[Nn] + 1),
                                            F(NB[Nn] + 2), so, vo, sbf);
  };

  // pre conv (no cutoff, no v in messages); inputs untouched.
  run_conv(0, in_p, in_v, 0, -1.f);
  run_update(0, in_s, s_cur, in_v, v_cur, in_p, p_cur);

  // 3 mid convs: lnorm -> conv(cutoff, with v) -> update
  for (int i = 0; i < 3; ++i) {
    run_norm(i, s_cur, v_cur);
    run_conv(1 + i, p_cur, v_cur, 1, 5.0f);
    run_update(1 + i, s_cur, s_cur, v_cur, v_cur, p_cur, p_cur);
  }

  // pnorm0 -> post conv -> update (final p goes straight to d_out)
  run_norm(3, s_cur, v_cur);
  run_conv(4, p_cur, v_cur, 0, -1.f);
  run_update(4, s_cur, s_cur, v_cur, v_cur, p_cur, out_p);

  // pnorm1 writes final s, v (and sbf for the edge head)
  run_norm(4, out_s, out_v);

  en_mlp<<<NNODE / 64, 128, 0, stream>>>(sbf, ep1t, F(60), ep2t, F(62), en);
  edge_out<<<(NEDGE + 255) / 256, 256, 0, stream>>>(eattr, en, ei, F(63), F(64),
                                                    F(65), F(66), out_e);
}
